// MLAAttention_87875030876602
// MI455X (gfx1250) — matrix operand order repaired and checked by reading, not compiled, not run on hardware
//
#include <hip/hip_runtime.h>
#include <hip/hip_bf16.h>
#include <math.h>

// ---------------------------------------------------------------------------
// MLA attention forward for MI455X (gfx1250, wave32, WMMA, async->LDS).
// One-time fp32->f16 conversion; whole chain runs f16 with fp32 accumulate
// via v_wmma_f32_16x16x32_f16. Tile staging uses GLOBAL_LOAD_ASYNC_TO_LDS_B128
// (ASYNCcnt) with double buffering. V is pre-transposed per head so every
// WMMA fragment load is a contiguous 16B-chunk load.
// ---------------------------------------------------------------------------

typedef __attribute__((ext_vector_type(16))) _Float16 v16h;
typedef __attribute__((ext_vector_type(8)))  _Float16 h8;
typedef __attribute__((ext_vector_type(8)))  float    v8f;

#define WMMA_F16(a, b, c) \
  __builtin_amdgcn_wmma_f32_16x16x32_f16(false, (a), false, (b), (short)0, (c), false, false)

// Problem constants
#define BS_       16
#define SEQ_      512
#define DMODEL_   2048
#define DOWN_     512
#define UP_       1024
#define NHEADS_   16
#define ROPE_     64
#define VHD_      64
#define QKD_      128

// ---- CDNA5 async global->LDS copy helpers (per-lane 16B) -------------------
__device__ __forceinline__ unsigned lds_off_of(const void* p) {
  // LDS aperture: addr[31:0] is the LDS byte offset (ISA 10.2 aperture map)
  return (unsigned)(uintptr_t)p;
}
__device__ __forceinline__ void async_b128(unsigned lds_byte_off, const void* gptr) {
  asm volatile("global_load_async_to_lds_b128 %0, %1, off"
               :: "v"(lds_byte_off), "v"(gptr) : "memory");
}
__device__ __forceinline__ void wait_async_0() {
  asm volatile("s_wait_asynccnt 0x0" ::: "memory");
}
__device__ __forceinline__ void wait_async_4() {
  asm volatile("s_wait_asynccnt 0x4" ::: "memory");
}

// ---- fragment loads (ISA 7.12.2 16-bit layouts, contiguous 16B chunks) -----
// A fragment: lane = row M (l16), a[i]=K(8*half+i), a[8+i]=K(16+8*half+i)
__device__ __forceinline__ v16h load_fragA(const _Float16* row, int half) {
  h8 lo = *(const h8*)(row + 8 * half);
  h8 hi = *(const h8*)(row + 16 + 8 * half);
  v16h a;
#pragma unroll
  for (int j = 0; j < 8; ++j) { a[j] = lo[j]; a[8 + j] = hi[j]; }
  return a;
}
// B fragment: lane = col N (l16), b[i]=K(8*half+i), b[8+i]=K(16+8*half+i)
__device__ __forceinline__ v16h load_fragB(const _Float16* row, int half) {
  h8 lo = *(const h8*)(row + 8 * half);
  h8 hi = *(const h8*)(row + 16 + 8 * half);
  v16h b;
#pragma unroll
  for (int j = 0; j < 8; ++j) { b[j] = lo[j]; b[8 + j] = hi[j]; }
  return b;
}

// ---------------------------------------------------------------------------
// fp32 -> f16 one-time conversion (vectorized x4)
// ---------------------------------------------------------------------------
typedef __attribute__((ext_vector_type(4))) _Float16 h4;
__global__ __launch_bounds__(256) void f32_to_f16(
    const float* __restrict__ src, _Float16* __restrict__ dst, int n4) {
  int i = blockIdx.x * blockDim.x + threadIdx.x;
  if (i >= n4) return;
  float4 v = *(const float4*)(src + (size_t)i * 4);
  h4 o; o.x = (_Float16)v.x; o.y = (_Float16)v.y; o.z = (_Float16)v.z; o.w = (_Float16)v.w;
  *(h4*)(dst + (size_t)i * 4) = o;
}

// ---------------------------------------------------------------------------
// GEMM:  C[M,N] = A[M,K] @ W[N,K]^T + bias[N]    (A, W f16; bias fp32)
// BM=BN=128, BK=32, 256 threads = 8 waves (2x4), each wave -> 64x32 of C.
// A and B tiles both staged row-major (M/N x K) via async b128 copies,
// double-buffered. OUT_F32 selects fp32 (final) vs f16 (chained) C.
// ---------------------------------------------------------------------------
#define BM 128
#define BN 128
#define BK 32
#define TLD 40   // tile row stride in f16 (32 + 8 pad; 80B = 16B-multiple)

template <bool OUT_F32>
__global__ __launch_bounds__(256) void gemm_wmma(
    int M, int N, int K,
    const _Float16* __restrict__ A,    // M x K
    const _Float16* __restrict__ W,    // N x K
    const float*    __restrict__ bias, // N
    void* __restrict__ Cv)             // M x N (f16 or f32)
{
  __shared__ _Float16 As[2][BM * TLD];
  __shared__ _Float16 Bs[2][BN * TLD];

  const int tid  = threadIdx.x;
  const int lane = tid & 31;
  const int wave = tid >> 5;
  const int wm   = wave >> 2;   // 0..1
  const int wn   = wave & 3;    // 0..3
  const int half = lane >> 4;
  const int l16  = lane & 15;

  const int bm = blockIdx.y * BM;
  const int bn = blockIdx.x * BN;

  v8f acc[4][2];
  v8f vzero = {};
#pragma unroll
  for (int i = 0; i < 4; ++i)
#pragma unroll
    for (int j = 0; j < 2; ++j) acc[i][j] = vzero;

  const unsigned asb[2] = { lds_off_of(&As[0][0]), lds_off_of(&As[1][0]) };
  const unsigned bsb[2] = { lds_off_of(&Bs[0][0]), lds_off_of(&Bs[1][0]) };

  auto stage = [&](int k0, int buf) {
#pragma unroll
    for (int p = 0; p < 2; ++p) {
      int ch = tid + p * 256;          // 0..511
      int r  = ch >> 2;                // tile row
      int c  = ch & 3;                 // 16B chunk in row
      async_b128(asb[buf] + (unsigned)(r * TLD + c * 8) * 2,
                 A + (size_t)(bm + r) * K + k0 + c * 8);
      int gn = bn + r; gn = gn < N ? gn : N - 1;   // branchless clamp (N=64 case)
      async_b128(bsb[buf] + (unsigned)(r * TLD + c * 8) * 2,
                 W + (size_t)gn * K + k0 + c * 8);
    }
  };

  const int niter = K / BK;
  stage(0, 0);

  for (int it = 0; it < niter; ++it) {
    const int buf = it & 1;
    if (it + 1 < niter) { stage((it + 1) * BK, buf ^ 1); wait_async_4(); }
    else                { wait_async_0(); }
    __syncthreads();

    v16h afrag[4];
#pragma unroll
    for (int tm = 0; tm < 4; ++tm)
      afrag[tm] = load_fragA(&As[buf][(wm * 64 + tm * 16 + l16) * TLD], half);
#pragma unroll
    for (int tn = 0; tn < 2; ++tn) {
      v16h bf = load_fragB(&Bs[buf][(wn * 32 + tn * 16 + l16) * TLD], half);
#pragma unroll
      for (int tm = 0; tm < 4; ++tm)
        acc[tm][tn] = WMMA_F16(afrag[tm], bf, acc[tm][tn]);
    }
    __syncthreads();
  }

  // epilogue (C/D layout: N = l16, M = r + 8*half)
#pragma unroll
  for (int tm = 0; tm < 4; ++tm)
#pragma unroll
    for (int tn = 0; tn < 2; ++tn) {
      int n = bn + wn * 32 + tn * 16 + l16;
      if (n >= N) continue;
      float bv = bias ? bias[n] : 0.f;
#pragma unroll
      for (int r = 0; r < 8; ++r) {
        int m = bm + wm * 64 + tm * 16 + r + half * 8;
        float v = acc[tm][tn][r] + bv;
        if (OUT_F32) ((float*)Cv)[(size_t)m * N + n] = v;
        else         ((_Float16*)Cv)[(size_t)m * N + n] = (_Float16)v;
      }
    }
}

// ---------------------------------------------------------------------------
// RoPE + concat into q,k (b,h,L,128) f16.
// Reference quirk: cos_pos = sin(pos*theta), sin_pos = cos(pos*theta).
// ---------------------------------------------------------------------------
__global__ __launch_bounds__(256) void build_qk(
    const _Float16* __restrict__ q_c,   // (bs*L, 1024)
    const _Float16* __restrict__ q_r,   // (bs*L, 1024)
    const _Float16* __restrict__ k_c,   // (bs*L, 1024)
    const _Float16* __restrict__ k_r,   // (bs*L, 64)
    _Float16* __restrict__ qbuf,        // (bs,H,L,128)
    _Float16* __restrict__ kbuf)
{
  int idx = blockIdx.x * blockDim.x + threadIdx.x;
  const int total = BS_ * NHEADS_ * SEQ_ * QKD_;
  if (idx >= total) return;
  int d = idx & 127;
  int l = (idx >> 7) & 511;
  int h = (idx >> 16) & 15;
  int b = idx >> 20;
  size_t tok = (size_t)b * SEQ_ + l;

  float qv, kv;
  if (d < 64) {
    qv = (float)q_c[tok * UP_ + h * 64 + d];
    kv = (float)k_c[tok * UP_ + h * 64 + d];
  } else {
    int dd = d - 64;
    int i  = dd >> 1;
    float theta = __powf(10000.f, -(float)(2 * i) / 64.f);
    float sn, cs;
    __sincosf((float)l * theta, &sn, &cs);
    float x  = (float)q_r[tok * UP_ + h * 64 + dd];
    float x2 = (dd & 1) ?  (float)q_r[tok * UP_ + h * 64 + dd - 1]
                        : -(float)q_r[tok * UP_ + h * 64 + dd + 1];
    qv = x * sn + x2 * cs;
    float y  = (float)k_r[tok * ROPE_ + dd];
    float y2 = (dd & 1) ?  (float)k_r[tok * ROPE_ + dd - 1]
                        : -(float)k_r[tok * ROPE_ + dd + 1];
    kv = y * sn + y2 * cs;
  }
  qbuf[idx] = (_Float16)qv;
  kbuf[idx] = (_Float16)kv;
}

// ---------------------------------------------------------------------------
// Per-head V transpose: v_c (bs*L, 1024) -> vT (bs,H,64,512), LDS-tiled so
// both the global read and the global write are coalesced.
// ---------------------------------------------------------------------------
__global__ __launch_bounds__(256) void transpose_v(
    const _Float16* __restrict__ v_c,   // (bs*L, 1024)
    _Float16* __restrict__ vT)          // (bs,H,64,512)
{
  __shared__ _Float16 t[64][72];        // 64x64 tile, +8 pad
  const int bh = blockIdx.x;            // b*16 + h
  const int b  = bh >> 4;
  const int h  = bh & 15;
  const int l0 = blockIdx.y * 64;

  for (int i = threadIdx.x; i < 64 * 64; i += 256) {
    int l = i >> 6, d = i & 63;         // read coalesced along d
    t[l][d] = v_c[((size_t)b * SEQ_ + l0 + l) * UP_ + h * VHD_ + d];
  }
  __syncthreads();
  for (int i = threadIdx.x; i < 64 * 64; i += 256) {
    int d = i >> 6, l = i & 63;         // write coalesced along l
    vT[((size_t)bh * VHD_ + d) * SEQ_ + l0 + l] = t[l][d];
  }
}

// ---------------------------------------------------------------------------
// Attention: one WG per (b, h, 64-row q tile). 256 threads = 8 waves.
// Q tile async-copied to LDS; S = scale*QK^T (WMMA, K frags from global);
// wave-per-row softmax; O = P (f16, LDS) @ V via WMMA with vT fragments.
// Dynamic LDS: 64*136*2 + 64*512*4 + 64*528*2 = 216064 B (< 320 KB/WG).
// ---------------------------------------------------------------------------
#define QT 64
#define QS_LD 136   // 128+8 (272B rows: 16B-multiple)
#define S_LD 512
#define P_LD 528

#define ATTN_SMEM_BYTES (QT*QS_LD*2 + QT*S_LD*4 + QT*P_LD*2)

__global__ __launch_bounds__(256) void mla_attention(
    const _Float16* __restrict__ qbuf,   // (bs,H,L,128)
    const _Float16* __restrict__ kbuf,   // (bs,H,L,128)
    const _Float16* __restrict__ vT,     // (bs,H,64,512)
    _Float16* __restrict__ attn_out)     // (bs*L, 1024)
{
  extern __shared__ char smem_raw[];
  _Float16* Qs = (_Float16*)smem_raw;
  float*    Ss = (float*)(smem_raw + QT * QS_LD * 2);
  _Float16* Ps = (_Float16*)(smem_raw + QT * QS_LD * 2 + QT * S_LD * 4);

  const int tid  = threadIdx.x;
  const int lane = tid & 31;
  const int wave = tid >> 5;
  const int half = lane >> 4;
  const int l16  = lane & 15;

  const int bh = blockIdx.x;
  const int b  = bh >> 4;
  const int h  = bh & 15;
  const int q0 = blockIdx.y * QT;

  const _Float16* Qg = qbuf + ((size_t)bh * SEQ_ + q0) * QKD_;
  const _Float16* Kg = kbuf + (size_t)bh * SEQ_ * QKD_;
  const _Float16* Vt = vT + (size_t)bh * VHD_ * SEQ_;

  // ---- 1) Q tile -> LDS via async b128 (1024 chunks, 4 per thread)
  {
    unsigned qs0 = lds_off_of(Qs);
#pragma unroll
    for (int p = 0; p < 4; ++p) {
      int ch = tid + p * 256;          // 0..1023
      int r  = ch >> 4;                // row (16 chunks of 16B per 256B row)
      int c  = ch & 15;
      async_b128(qs0 + (unsigned)(r * QS_LD + c * 8) * 2,
                 Qg + (size_t)r * QKD_ + c * 8);
    }
    wait_async_0();
  }
  __syncthreads();

  const float scale = 0.07216878364870323f;   // 1/sqrt(192)

  // ---- 2) scores: 4 M-tiles x 32 N-tiles; wave owns N-tiles [wave*4,+4)
  for (int mt = 0; mt < 4; ++mt) {
    v16h afrag[4];
#pragma unroll
    for (int kc = 0; kc < 4; ++kc)
      afrag[kc] = load_fragA(&Qs[(mt * 16 + l16) * QS_LD + kc * 32], half);
#pragma unroll
    for (int nt4 = 0; nt4 < 4; ++nt4) {
      int nt   = wave * 4 + nt4;
      int nkey = nt * 16 + l16;
      v8f acc = {};
#pragma unroll
      for (int kc = 0; kc < 4; ++kc) {
        v16h bf = load_fragB(Kg + (size_t)nkey * QKD_ + kc * 32, half);
        acc = WMMA_F16(afrag[kc], bf, acc);
      }
#pragma unroll
      for (int r = 0; r < 8; ++r) {
        int m = mt * 16 + r + half * 8;
        Ss[m * S_LD + nt * 16 + l16] = acc[r] * scale;
      }
    }
  }
  __syncthreads();

  // ---- 3) softmax, wave per row (8 rows per wave)
#pragma unroll
  for (int rr = 0; rr < 8; ++rr) {
    int r = wave * 8 + rr;
    float* row = &Ss[r * S_LD];
    float mx = -3.402823466e38f;
    for (int c = lane; c < SEQ_; c += 32) mx = fmaxf(mx, row[c]);
#pragma unroll
    for (int off = 16; off >= 1; off >>= 1) mx = fmaxf(mx, __shfl_xor(mx, off, 32));
    float sum = 0.f;
    for (int c = lane; c < SEQ_; c += 32) {
      float e = __expf(row[c] - mx);
      row[c] = e;
      sum += e;
    }
#pragma unroll
    for (int off = 16; off >= 1; off >>= 1) sum += __shfl_xor(sum, off, 32);
    float inv = 1.f / sum;
    for (int c = lane; c < SEQ_; c += 32)
      Ps[r * P_LD + c] = (_Float16)(row[c] * inv);
  }
  __syncthreads();

  // ---- 4) O = P @ V : 4x4 tiles, 2 per wave; V frags contiguous from vT
  int mt = wave & 3;
#pragma unroll
  for (int tp = 0; tp < 2; ++tp) {
    int nt = (wave >> 2) * 2 + tp;
    int n  = nt * 16 + l16;
    v8f acc = {};
    for (int kc = 0; kc < 16; ++kc) {
      v16h a  = load_fragA(&Ps[(mt * 16 + l16) * P_LD + kc * 32], half);
      v16h bf = load_fragB(Vt + (size_t)n * SEQ_ + kc * 32, half);
      acc = WMMA_F16(a, bf, acc);
    }
#pragma unroll
    for (int r = 0; r < 8; ++r) {
      int m = mt * 16 + r + half * 8;
      attn_out[((size_t)b * SEQ_ + q0 + m) * UP_ + h * VHD_ + n] = (_Float16)acc[r];
    }
  }
}

// ---------------------------------------------------------------------------
// Host-side orchestration
// ---------------------------------------------------------------------------
extern "C" void kernel_launch(void* const* d_in, const int* in_sizes, int n_in,
                              void* d_out, int out_size, void* d_ws, size_t ws_size,
                              hipStream_t stream) {
  const float* h_in      = (const float*)d_in[0];
  const float* W_down_kv = (const float*)d_in[1];
  const float* b_down_kv = (const float*)d_in[2];
  const float* W_up_k    = (const float*)d_in[3];
  const float* b_up_k    = (const float*)d_in[4];
  const float* W_up_v    = (const float*)d_in[5];
  const float* b_up_v    = (const float*)d_in[6];
  const float* W_down_q  = (const float*)d_in[7];
  const float* b_down_q  = (const float*)d_in[8];
  const float* W_up_q    = (const float*)d_in[9];
  const float* b_up_q    = (const float*)d_in[10];
  const float* W_qr      = (const float*)d_in[11];
  const float* b_qr      = (const float*)d_in[12];
  const float* W_kr      = (const float*)d_in[13];
  const float* b_kr      = (const float*)d_in[14];
  const float* W_fc      = (const float*)d_in[15];
  const float* b_fc      = (const float*)d_in[16];
  float* out = (float*)d_out;
  (void)in_sizes; (void)n_in; (void)out_size; (void)ws_size;

  const size_t TOK = (size_t)BS_ * SEQ_;   // 8192

  // f16 workspace layout (~232 MiB)
  _Float16* ws = (_Float16*)d_ws;
  size_t off = 0;
  auto alloc = [&](size_t n) { _Float16* p = ws + off; off += n; return p; };
  _Float16* h16    = alloc(TOK * DMODEL_);
  _Float16* wdkv   = alloc((size_t)DOWN_ * DMODEL_);
  _Float16* wupk   = alloc((size_t)UP_ * DOWN_);
  _Float16* wupv   = alloc((size_t)UP_ * DOWN_);
  _Float16* wdq    = alloc((size_t)DOWN_ * DMODEL_);
  _Float16* wupq   = alloc((size_t)UP_ * DOWN_);
  _Float16* wqr    = alloc((size_t)UP_ * DOWN_);
  _Float16* wkr    = alloc((size_t)ROPE_ * DMODEL_);
  _Float16* wfc    = alloc((size_t)DMODEL_ * UP_);
  _Float16* c_kv   = alloc(TOK * DOWN_);
  _Float16* c_q    = alloc(TOK * DOWN_);
  _Float16* k_c    = alloc(TOK * UP_);
  _Float16* v_c    = alloc(TOK * UP_);
  _Float16* q_c    = alloc(TOK * UP_);
  _Float16* q_r    = alloc(TOK * UP_);
  _Float16* k_r    = alloc(TOK * ROPE_);
  _Float16* qbuf   = alloc((size_t)BS_ * NHEADS_ * SEQ_ * QKD_);
  _Float16* kbuf   = alloc((size_t)BS_ * NHEADS_ * SEQ_ * QKD_);
  _Float16* vTb    = alloc((size_t)BS_ * NHEADS_ * VHD_ * SEQ_);
  _Float16* attn_o = alloc(TOK * UP_);

  auto cvt = [&](const float* s, _Float16* d, size_t n) {
    int n4 = (int)(n / 4);
    f32_to_f16<<<(n4 + 255) / 256, 256, 0, stream>>>(s, d, n4);
  };
  cvt(h_in,      h16,  TOK * DMODEL_);
  cvt(W_down_kv, wdkv, (size_t)DOWN_ * DMODEL_);
  cvt(W_up_k,    wupk, (size_t)UP_ * DOWN_);
  cvt(W_up_v,    wupv, (size_t)UP_ * DOWN_);
  cvt(W_down_q,  wdq,  (size_t)DOWN_ * DMODEL_);
  cvt(W_up_q,    wupq, (size_t)UP_ * DOWN_);
  cvt(W_qr,      wqr,  (size_t)UP_ * DOWN_);
  cvt(W_kr,      wkr,  (size_t)ROPE_ * DMODEL_);
  cvt(W_fc,      wfc,  (size_t)DMODEL_ * UP_);

  auto gemm16 = [&](int M, int N, int K, const _Float16* A, const _Float16* W,
                    const float* bias, _Float16* C) {
    dim3 grid((N + BN - 1) / BN, (M + BM - 1) / BM);
    gemm_wmma<false><<<grid, 256, 0, stream>>>(M, N, K, A, W, bias, (void*)C);
  };

  gemm16((int)TOK, DOWN_, DMODEL_, h16,  wdkv, b_down_kv, c_kv);
  gemm16((int)TOK, UP_,   DOWN_,   c_kv, wupk, b_up_k,    k_c);
  gemm16((int)TOK, UP_,   DOWN_,   c_kv, wupv, b_up_v,    v_c);
  gemm16((int)TOK, DOWN_, DMODEL_, h16,  wdq,  b_down_q,  c_q);
  gemm16((int)TOK, UP_,   DOWN_,   c_q,  wupq, b_up_q,    q_c);
  gemm16((int)TOK, UP_,   DOWN_,   c_q,  wqr,  b_qr,      q_r);
  gemm16((int)TOK, ROPE_, DMODEL_, h16,  wkr,  b_kr,      k_r);

  {
    const int total = BS_ * NHEADS_ * SEQ_ * QKD_;
    build_qk<<<(total + 255) / 256, 256, 0, stream>>>(q_c, q_r, k_c, k_r, qbuf, kbuf);
  }
  {
    dim3 grid(BS_ * NHEADS_, SEQ_ / 64);
    transpose_v<<<grid, 256, 0, stream>>>(v_c, vTb);
  }
  {
    dim3 grid(BS_ * NHEADS_, SEQ_ / QT);
    mla_attention<<<grid, 256, ATTN_SMEM_BYTES, stream>>>(qbuf, kbuf, vTb, attn_o);
  }
  {
    dim3 grid((DMODEL_ + BN - 1) / BN, ((int)TOK + BM - 1) / BM);
    gemm_wmma<true><<<grid, 256, 0, stream>>>((int)TOK, DMODEL_, UP_,
                                              attn_o, wfc, b_fc, (void*)out);
  }
}
